// SAGE_72791105732746
// MI455X (gfx1250) — compile-verified, matrix operand order checked by reading, not run on hardware
//
#include <hip/hip_runtime.h>
#include <hip/hip_bf16.h>

typedef __attribute__((ext_vector_type(16))) _Float16 v16h;
typedef __attribute__((ext_vector_type(8)))  float    v8f;

#define HD   128
#define LDP  (HD + 8)          // padded LDS row stride (halves) -> conflict-free column reads
#define EPSV 1e-5f

// ---------------------------------------------------------------- utilities
__global__ __launch_bounds__(256) void zero_f32(float* __restrict__ p, long long n) {
  long long i = (long long)blockIdx.x * blockDim.x + threadIdx.x;
  long long stride = (long long)gridDim.x * blockDim.x;
  for (; i < n; i += stride) p[i] = 0.0f;
}

__global__ __launch_bounds__(256) void count_deg(const int* __restrict__ dst,
                                                 float* __restrict__ cnt, int E) {
  int i = blockIdx.x * blockDim.x + threadIdx.x;
  if (i < E) atomicAdd(&cnt[dst[i]], 1.0f);
}

// one edge handled by 32 lanes, each lane moves a float4 (coalesced 512B per edge-row)
__global__ __launch_bounds__(256) void scatter_feat(const float* __restrict__ x,
                                                    const int* __restrict__ src,
                                                    const int* __restrict__ dst,
                                                    float* __restrict__ s, int E) {
  long long idx = (long long)blockIdx.x * blockDim.x + threadIdx.x;
  long long total = (long long)E * 32;
  if (idx >= total) return;
  int e    = (int)(idx >> 5);
  int lane = (int)(idx & 31);
  int sr = src[e], ds = dst[e];
  const float4 v = *(const float4*)(x + (long long)sr * HD + lane * 4);
  float* o = s + (long long)ds * HD + lane * 4;
  atomicAdd(o + 0, v.x);
  atomicAdd(o + 1, v.y);
  atomicAdd(o + 2, v.z);
  atomicAdd(o + 3, v.w);
}

// ------------------------------------------------- weight pre-pack (f32 -> f16 B-fragments)
// P layout: [n_tile 0..7][kk 0..3][lane 0..31][16 halves, contiguous]
// Fragment element 2i   = W[k][n0+n], 2i+1 = W[k+1][n0+n]
// with k = kk*32 + (i<4 ? 2i : 16+2(i-4)) + 8*(lane>>4), n = lane&15  (ISA 16-bit B layout)
__global__ __launch_bounds__(256) void pack_w(const float* __restrict__ W,
                                              _Float16* __restrict__ P) {
  int idx = blockIdx.x * 256 + threadIdx.x;   // 0 .. 16383
  if (idx >= HD * HD) return;
  int j    = idx & 15;            // half index within lane chunk
  int lane = (idx >> 4) & 31;
  int kk   = ((idx >> 9) & 3) * 32;
  int t    = idx >> 11;           // output-column tile 0..7
  int n    = lane & 15;
  int hi   = lane >> 4;
  int i    = j >> 1, off = j & 1;
  int k    = kk + ((i < 4) ? (2 * i) : (16 + 2 * (i - 4))) + hi * 8 + off;
  P[idx] = (_Float16)W[k * HD + t * 16 + n];
}

// ------------------------------------------------- fused aggr + dual WMMA GEMM
// out[16-row tile] = (s/max(cnt,1)) @ Wl + x @ Wr + bl
// 256 threads = 8 waves; wave w computes output columns [16w, 16w+16)
__global__ __launch_bounds__(256) void sage_gemm(
    const float*    __restrict__ x,    // [N,128]
    const float*    __restrict__ s,    // [N,128] neighbor sums
    const float*    __restrict__ cnt,  // [N]
    const _Float16* __restrict__ Pl,   // packed Wl fragments
    const _Float16* __restrict__ Pr,   // packed Wr fragments
    const float*    __restrict__ bl,   // [128]
    float*          __restrict__ out,  // [N,128]
    int N) {
  __shared__ _Float16 Ax[16][LDP];
  __shared__ _Float16 Aa[16][LDP];

  const int r0  = blockIdx.x * 16;
  const int tid = threadIdx.x;

  // stage A tiles: x rows and mean-aggregated rows, f32 -> f16
  for (int i = tid; i < 16 * HD; i += 256) {
    int row = i >> 7, col = i & 127;
    int gr = r0 + row;
    if (gr >= N) gr = N - 1;
    long long g = (long long)gr * HD + col;
    float inv = 1.0f / fmaxf(cnt[gr], 1.0f);
    Ax[row][col] = (_Float16)x[g];
    Aa[row][col] = (_Float16)(s[g] * inv);
  }
  __syncthreads();

  const int wave = tid >> 5;     // 0..7 -> output column tile
  const int lane = tid & 31;
  const int n0   = wave * 16;
  const int n    = lane & 15;    // output column within tile
  const int hi   = lane >> 4;    // half-wave: K offset +8, C rows +8
  const int m    = lane & 15;    // A row index

  v8f c = {};
  #pragma unroll
  for (int kt = 0; kt < 4; ++kt) {
    const int kk = kt * 32;
    // A fragments: two contiguous 8-half (16B) chunks per matrix -> ds_load_b128
    v16h aa, ax;
    #pragma unroll
    for (int i = 0; i < 8; ++i) {
      int kl = (i < 4 ? (i * 2) : (16 + (i - 4) * 2)) + hi * 8;
      int k  = kk + kl;
      aa[2*i]   = Aa[m][k];
      aa[2*i+1] = Aa[m][k+1];
      ax[2*i]   = Ax[m][k];
      ax[2*i+1] = Ax[m][k+1];
    }
    // B fragments: one 32B aligned v16h load each -> 2x global_load_b128
    const long long pb = (((long long)wave * 4 + kt) * 32 + lane) * 16;
    v16h bwl = *(const v16h*)(Pl + pb);
    v16h bwr = *(const v16h*)(Pr + pb);

    c = __builtin_amdgcn_wmma_f32_16x16x32_f16(false, aa, false, bwl, (short)0, c, false, false);
    c = __builtin_amdgcn_wmma_f32_16x16x32_f16(false, ax, false, bwr, (short)0, c, false, false);
  }

  const float bias = bl[n0 + n];
  #pragma unroll
  for (int r = 0; r < 8; ++r) {
    int row = r0 + r + hi * 8;   // C/D layout: VGPR r -> M = r + 8*hi, N = lane&15
    if (row < N) out[(long long)row * HD + n0 + n] = c[r] + bias;
  }
}

// ---------------------------------------------------------------- BN + ReLU
__global__ __launch_bounds__(128) void col_stats(const float* __restrict__ h,
                                                 float* __restrict__ stats, int N) {
  int col = threadIdx.x;  // 0..127
  float sum = 0.0f, sq = 0.0f;
  for (int r = blockIdx.x; r < N; r += gridDim.x) {
    float v = h[(long long)r * HD + col];
    sum += v;
    sq  += v * v;
  }
  atomicAdd(&stats[col], sum);
  atomicAdd(&stats[HD + col], sq);
}

__global__ __launch_bounds__(256) void bn_relu(const float* __restrict__ h,
                                               const float* __restrict__ stats,
                                               const float* __restrict__ g,
                                               const float* __restrict__ b,
                                               float* __restrict__ out, int N) {
  long long i = (long long)blockIdx.x * blockDim.x + threadIdx.x;
  long long total = (long long)N * HD;
  long long stride = (long long)gridDim.x * blockDim.x;
  const float invN = 1.0f / (float)N;
  for (; i < total; i += stride) {
    int col = (int)(i & 127);
    float mu  = stats[col] * invN;
    float var = stats[HD + col] * invN - mu * mu;
    float y = (h[i] - mu) * rsqrtf(var + EPSV) * g[col] + b[col];
    out[i] = fmaxf(y, 0.0f);
  }
}

// ---------------------------------------------------------------- driver
extern "C" void kernel_launch(void* const* d_in, const int* in_sizes, int n_in,
                              void* d_out, int out_size, void* d_ws, size_t ws_size,
                              hipStream_t stream) {
  (void)n_in; (void)out_size; (void)ws_size;

  const float* x   = (const float*)d_in[0];
  const int*   esr = (const int*)  d_in[1];
  const int*   eds = (const int*)  d_in[2];
  const float* Wl0 = (const float*)d_in[3];
  const float* bl0 = (const float*)d_in[4];
  const float* Wr0 = (const float*)d_in[5];
  const float* g0  = (const float*)d_in[6];
  const float* b0  = (const float*)d_in[7];
  const float* Wl1 = (const float*)d_in[8];
  const float* bl1 = (const float*)d_in[9];
  const float* Wr1 = (const float*)d_in[10];
  const float* g1  = (const float*)d_in[11];
  const float* b1  = (const float*)d_in[12];
  const float* Wl2 = (const float*)d_in[13];
  const float* bl2 = (const float*)d_in[14];
  const float* Wr2 = (const float*)d_in[15];

  const int N = in_sizes[0] / HD;
  const int E = in_sizes[1];

  float* sbuf  = (float*)d_ws;                        // [N*128] neighbor sums
  float* tbuf  = sbuf + (size_t)N * HD;               // [N*128] pre-BN activations
  float* abuf  = tbuf + (size_t)N * HD;               // [N*128] post-BN activations
  float* cnt   = abuf + (size_t)N * HD;               // [N]
  float* stats = cnt + N;                             // [256]
  _Float16* pw = (_Float16*)(stats + 2 * HD);         // 6 x 16384 packed f16 weights
  _Float16* pWl0 = pw;
  _Float16* pWr0 = pWl0 + HD * HD;
  _Float16* pWl1 = pWr0 + HD * HD;
  _Float16* pWr1 = pWl1 + HD * HD;
  _Float16* pWl2 = pWr1 + HD * HD;
  _Float16* pWr2 = pWl2 + HD * HD;

  const int gemm_blocks = (N + 15) / 16;
  const long long sc_total = (long long)E * 32;
  const int sc_blocks = (int)((sc_total + 255) / 256);
  const int pk_blocks = (HD * HD + 255) / 256;

  // pre-pack all six weight matrices into WMMA B-fragment order (f16)
  pack_w<<<pk_blocks, 256, 0, stream>>>(Wl0, pWl0);
  pack_w<<<pk_blocks, 256, 0, stream>>>(Wr0, pWr0);
  pack_w<<<pk_blocks, 256, 0, stream>>>(Wl1, pWl1);
  pack_w<<<pk_blocks, 256, 0, stream>>>(Wr1, pWr1);
  pack_w<<<pk_blocks, 256, 0, stream>>>(Wl2, pWl2);
  pack_w<<<pk_blocks, 256, 0, stream>>>(Wr2, pWr2);

  // degrees (edge-set invariant across layers: compute once)
  zero_f32<<<256, 256, 0, stream>>>(cnt, N);
  count_deg<<<(E + 255) / 256, 256, 0, stream>>>(eds, cnt, E);

  // ---- layer 0: SAGE(D->H) + BN + ReLU
  zero_f32<<<2048, 256, 0, stream>>>(sbuf, (long long)N * HD);
  scatter_feat<<<sc_blocks, 256, 0, stream>>>(x, esr, eds, sbuf, E);
  sage_gemm<<<gemm_blocks, 256, 0, stream>>>(x, sbuf, cnt, pWl0, pWr0, bl0, tbuf, N);
  zero_f32<<<1, 256, 0, stream>>>(stats, 2 * HD);
  col_stats<<<512, 128, 0, stream>>>(tbuf, stats, N);
  bn_relu<<<2048, 256, 0, stream>>>(tbuf, stats, g0, b0, abuf, N);

  // ---- layer 1: SAGE(H->H) + BN + ReLU
  zero_f32<<<2048, 256, 0, stream>>>(sbuf, (long long)N * HD);
  scatter_feat<<<sc_blocks, 256, 0, stream>>>(abuf, esr, eds, sbuf, E);
  sage_gemm<<<gemm_blocks, 256, 0, stream>>>(abuf, sbuf, cnt, pWl1, pWr1, bl1, tbuf, N);
  zero_f32<<<1, 256, 0, stream>>>(stats, 2 * HD);
  col_stats<<<512, 128, 0, stream>>>(tbuf, stats, N);
  bn_relu<<<2048, 256, 0, stream>>>(tbuf, stats, g1, b1, abuf, N);

  // ---- layer 2: SAGE(H->O), no BN/ReLU, straight to d_out
  zero_f32<<<2048, 256, 0, stream>>>(sbuf, (long long)N * HD);
  scatter_feat<<<sc_blocks, 256, 0, stream>>>(abuf, esr, eds, sbuf, E);
  sage_gemm<<<gemm_blocks, 256, 0, stream>>>(abuf, sbuf, cnt, pWl2, pWr2, bl2,
                                             (float*)d_out, N);
}
